// MentalHealthGNN_18863496364560
// MI455X (gfx1250) — matrix operand order repaired and checked by reading, not compiled, not run on hardware
//
#include <hip/hip_runtime.h>
#include <math.h>

// ---------------------------------------------------------------------------
// Problem constants (from the reference, compile-time)
// Node types in jax-sorted order: document, medical_concept, symptom_category, word
// All GEMM dims padded: M to 16, K to 32, N to 64 (N already 128/384).
// ---------------------------------------------------------------------------
#define NTYPES 4
#define ETYPES_N 8
#define NL 2
#define NDOC 20000
#define NTOTP 75072           // padded: 20000 + 5008 + 64 + 50000
#define KETOT 185050          // sum of per-edge-type source-node counts (unpadded)
#define ETOT 1360000          // total edges
#define AH_MAX 16000000       // max A-halves: word proj 50000*320
#define WT_MAX 98304          // max W-halves: 128*768

static const int NT_CNT[NTYPES]  = {20000, 5000, 50, 50000};
static const int NT_PAD[NTYPES]  = {20000, 5008, 64, 50000};     // rows padded to 16
static const int NT_DIM[NTYPES]  = {768, 128, 64, 300};
static const int NT_KPAD[NTYPES] = {768, 128, 64, 320};          // K padded to 32
static const int NT_OFFP[NTYPES] = {0, 20000, 25008, 25072};     // padded row offsets
static const int NT_XIDX[NTYPES] = {0, 2, 3, 1};                 // d_in index of x_<type>

// Edge types sorted: belongs_to, co_occurs, contains, maps_to, related_to,
//                    rev_belongs_to, rev_contains, rev_maps_to
static const int ET_SRC[ETYPES_N] = {1, 3, 0, 3, 1, 2, 3, 1};
static const int ET_DST[ETYPES_N] = {2, 3, 3, 1, 1, 1, 0, 3};
static const int ET_CNT[ETYPES_N] = {5000, 300000, 400000, 100000, 50000, 5000, 400000, 100000};
static const int ET_EIIDX[ETYPES_N] = {7, 5, 4, 6, 8, 11, 9, 10};

typedef _Float16 half8  __attribute__((ext_vector_type(8)));
typedef _Float16 half16 __attribute__((ext_vector_type(16)));
typedef float    float8 __attribute__((ext_vector_type(8)));

__device__ __forceinline__ float gelu_f(float x) {
    return 0.5f * x * (1.0f + erff(x * 0.70710678118654752f));
}
__device__ __forceinline__ int encF(float f) {
    int i = __float_as_int(f);
    return (i >= 0) ? i : (i ^ 0x7FFFFFFF);
}
__device__ __forceinline__ float decF(int i) {
    return __int_as_float((i >= 0) ? i : (i ^ 0x7FFFFFFF));
}

// ---------------------------------------------------------------------------
// A conversion: f32 [Mreal x Ksrc] -> f16 row-major [Mpad x Kdst], zero pad,
// optional exact-GELU fused (HGT out-projection input).
// ---------------------------------------------------------------------------
__global__ void conv_a_kernel(const float* __restrict__ src, _Float16* __restrict__ dst,
                              int Mreal, int Mpad, int Ksrc, int Kdst, int dogelu)
{
    int idx = blockIdx.x * 256 + threadIdx.x;
    if (idx >= Mpad * Kdst) return;
    int r = idx / Kdst, c = idx - r * Kdst;
    float v = (r < Mreal && c < Ksrc) ? src[(size_t)r * Ksrc + c] : 0.f;
    if (dogelu) v = gelu_f(v);
    dst[idx] = (_Float16)v;
}

// W conversion: f32 row-major [Ksrc x N] -> f16 COLUMN-major [N x Kdst], zero pad.
__global__ void conv_w_kernel(const float* __restrict__ src, _Float16* __restrict__ dst,
                              int Ksrc, int Kdst, int N)
{
    int idx = blockIdx.x * 256 + threadIdx.x;
    if (idx >= N * Kdst) return;
    int n = idx / Kdst, k = idx - n * Kdst;
    dst[idx] = (_Float16)((k < Ksrc) ? src[(size_t)k * N + n] : 0.f);
}

// ---------------------------------------------------------------------------
// WMMA GEMM: C[m,ldc] = epi(Ah[MxK] @ Wt^T + bias); Ah row-major f16,
// Wt column-major f16. One wave computes a 16x64 strip: A fragment reused
// across 4 B tiles -> 4 v_wmma per 32-wide K step, all loads b128, no guards.
//   mode 0: +bias   mode 1: +bias,ReLU
//   mode 2: C = ReLU( sg*(acc+bias) + (1-sg)*hprev ), sg=sigmoid(*skipw)
// ---------------------------------------------------------------------------
__global__ void gemm_wmma_kernel(const _Float16* __restrict__ Ah,
                                 const _Float16* __restrict__ Wt,
                                 const float* __restrict__ bias,
                                 float* __restrict__ C,
                                 int M, int N, int K, int ldc, int mode,
                                 const float* __restrict__ skipw,
                                 const float* __restrict__ hprev)
{
    const int lane = threadIdx.x & 31;
    const int wave = threadIdx.x >> 5;
    const int row0 = (blockIdx.y * 4 + wave) * 16;
    if (row0 >= M) return;                       // wave-uniform
    const int col0 = blockIdx.x * 64;
    const int kh  = lane >> 4;
    const int l15 = lane & 15;

    // A fragment (16x32 f16): lane holds row row0+l15; two contiguous 8-half
    // runs at K offsets kh*16 and kh*16+8 (contiguous, matches B).
    const _Float16* ap = Ah + (size_t)(row0 + l15) * K + (kh << 4);
    // B fragment (32x16 f16): lane holds col; one contiguous 16-half run at
    // K offset kh*16 in column-major Wt.
    const _Float16* bp = Wt + (size_t)(col0 + l15) * K + (kh << 4);
    const size_t bstride = (size_t)16 * K;

    float8 acc[4] = {};
    for (int kk = 0; kk < K; kk += 32) {
        half8 alo = *(const half8*)(ap);
        half8 ahi = *(const half8*)(ap + 8);
        ap += 32;
        half16 af = __builtin_shufflevector(alo, ahi,
                        0,1,2,3,4,5,6,7,8,9,10,11,12,13,14,15);
        half16 b0 = *(const half16*)(bp);
        half16 b1 = *(const half16*)(bp + bstride);
        half16 b2 = *(const half16*)(bp + 2 * bstride);
        half16 b3 = *(const half16*)(bp + 3 * bstride);
        bp += 32;
        acc[0] = __builtin_amdgcn_wmma_f32_16x16x32_f16(false, af, false, b0, (short)0, acc[0], false, false);
        acc[1] = __builtin_amdgcn_wmma_f32_16x16x32_f16(false, af, false, b1, (short)0, acc[1], false, false);
        acc[2] = __builtin_amdgcn_wmma_f32_16x16x32_f16(false, af, false, b2, (short)0, acc[2], false, false);
        acc[3] = __builtin_amdgcn_wmma_f32_16x16x32_f16(false, af, false, b3, (short)0, acc[3], false, false);
    }

    const float sg = (mode == 2) ? 1.0f / (1.0f + expf(-skipw[0])) : 0.f;
#pragma unroll
    for (int t = 0; t < 4; ++t) {
        const int col = col0 + t * 16 + l15;
        const float bb = bias[col];
#pragma unroll
        for (int v = 0; v < 8; ++v) {
            const int m = row0 + kh * 8 + v;     // C layout: lanes 0-15 M=v, 16-31 M=v+8
            float val = acc[t][v] + bb;
            if (mode == 1) {
                val = fmaxf(val, 0.f);
            } else if (mode == 2) {
                float prev = hprev[(size_t)m * 128 + col];
                val = sg * val + (1.f - sg) * prev;
                val = fmaxf(val, 0.f);
            }
            C[(size_t)m * ldc + col] = val;
        }
    }
}

// ---------------------------------------------------------------------------
// Per-head 16x16 relation transforms: ke = k @ a_k[h], ve = v @ a_v[h]
// kqv rows: [k(128) | q(128) | v(128)]
// ---------------------------------------------------------------------------
__global__ void kvrel_kernel(const float* __restrict__ kqv, int nodeOff,
                             const float* __restrict__ ak, const float* __restrict__ av,
                             float* __restrict__ ke, float* __restrict__ ve, int ns)
{
    int idx = blockIdx.x * 256 + threadIdx.x;
    if (idx >= ns * 128) return;
    int n = idx >> 7, hd = idx & 127, h = hd >> 4, d2 = hd & 15;
    const float* krow = kqv + (size_t)(nodeOff + n) * 384 + h * 16;
    const float* vrow = krow + 256;
    const float* akh = ak + h * 256 + d2;
    const float* avh = av + h * 256 + d2;
    float sk = 0.f, sv = 0.f;
#pragma unroll
    for (int d = 0; d < 16; ++d) {
        sk += krow[d] * akh[d * 16];
        sv += vrow[d] * avh[d * 16];
    }
    ke[(size_t)n * 128 + hd] = sk;
    ve[(size_t)n * 128 + hd] = sv;
}

__global__ void init_nodes_kernel(int* __restrict__ nmaxI, float* __restrict__ nsum,
                                  float* __restrict__ agg)
{
    int i = blockIdx.x * 256 + threadIdx.x;
    if (i < NTOTP * 8) { nmaxI[i] = (int)0x80000000; nsum[i] = 0.f; }
    if (i < NTOTP * 128) agg[i] = 0.f;
}

// Pass 1: alpha = <q[dst], ke[src]> * p[h] / sqrt(D); segment max via int atomicMax
__global__ void edge_pass1_kernel(const int* __restrict__ ei, int E,
                                  const float* __restrict__ kqv, int dstOff,
                                  const float* __restrict__ ke,
                                  const float* __restrict__ p,
                                  float* __restrict__ alpha, int* __restrict__ nmaxI)
{
    int gid = blockIdx.x * 256 + threadIdx.x;
    if (gid >= E * 8) return;
    int e = gid >> 3, h = gid & 7;
    int si = ei[e], di = ei[E + e];
    const float* qrow  = kqv + (size_t)(dstOff + di) * 384 + 128 + h * 16;
    const float* kerow = ke  + (size_t)si * 128 + h * 16;
    float a = 0.f;
#pragma unroll
    for (int d = 0; d < 16; ++d) a += qrow[d] * kerow[d];
    a *= p[h] * 0.25f;                       // 1/sqrt(16)
    alpha[gid] = a;
    atomicMax(&nmaxI[(size_t)(dstOff + di) * 8 + h], encF(a));
}

// Pass 2: w = exp(alpha - max); segment sum via f32 atomicAdd
__global__ void edge_pass2_kernel(const int* __restrict__ ei, int E, int dstOff,
                                  float* __restrict__ alpha,
                                  const int* __restrict__ nmaxI,
                                  float* __restrict__ nsum)
{
    int gid = blockIdx.x * 256 + threadIdx.x;
    if (gid >= E * 8) return;
    int e = gid >> 3, h = gid & 7;
    int di = ei[E + e];
    int g8 = (dstOff + di) * 8 + h;
    float w = expf(alpha[gid] - decF(nmaxI[g8]));
    alpha[gid] = w;
    atomicAdd(&nsum[g8], w);
}

// Pass 3: agg[dst] += (w/den) * ve[src]; one wave (32 lanes x 4 cols) per edge
__global__ void edge_pass3_kernel(const int* __restrict__ ei, int E, int dstOff,
                                  const float* __restrict__ ve,
                                  const float* __restrict__ alpha,
                                  const float* __restrict__ nsum,
                                  float* __restrict__ agg)
{
    int eb = blockIdx.x * 8 + (threadIdx.x >> 5);
    if (eb >= E) return;
    int lane = threadIdx.x & 31;
    int si = ei[eb], di = ei[E + eb];
    size_t gd = (size_t)(dstOff + di);
#pragma unroll
    for (int j = 0; j < 4; ++j) {
        int c = lane + j * 32;
        int h = c >> 4;
        float wn = alpha[(size_t)eb * 8 + h] / (nsum[gd * 8 + h] + 1e-16f);
        atomicAdd(&agg[gd * 128 + c], wn * ve[(size_t)si * 128 + c]);
    }
}

// doc[:, :128] = h_document
__global__ void copy_doc_kernel(const float* __restrict__ h, float* __restrict__ doc)
{
    int i = blockIdx.x * 256 + threadIdx.x;
    if (i >= NDOC * 128) return;
    int n = i >> 7, c = i & 127;
    doc[(size_t)n * 256 + c] = h[i];
}

// logits = hid @ W2 + b2  (128 -> 2, too narrow for WMMA tiles)
__global__ void cls2_kernel(const float* __restrict__ hid, const float* __restrict__ w,
                            const float* __restrict__ b, float* __restrict__ out)
{
    int i = blockIdx.x * 256 + threadIdx.x;
    if (i >= NDOC * 2) return;
    int n = i >> 1, c = i & 1;
    float s = b[c];
    const float* hr = hid + (size_t)n * 128;
#pragma unroll 16
    for (int k = 0; k < 128; ++k) s += hr[k] * w[k * 2 + c];
    out[i] = s;
}

// ---------------------------------------------------------------------------
// Host-side orchestration
// ---------------------------------------------------------------------------
extern "C" void kernel_launch(void* const* d_in, const int* in_sizes, int n_in,
                              void* d_out, int out_size, void* d_ws, size_t ws_size,
                              hipStream_t stream)
{
    (void)in_sizes; (void)n_in; (void)out_size; (void)ws_size;
    auto F = [&](int i) { return (const float*)d_in[i]; };

    // ---- params: jax pytree leaves in sorted-key order, after 12 array inputs
    int pi = 12;
    const float* cls1_b = F(pi++); const float* cls1_w = F(pi++);
    const float* cls2_b = F(pi++); const float* cls2_w = F(pi++);
    struct LayerP {
        const float *a_k[ETYPES_N], *a_v[ETYPES_N];
        const float *kqv_b[NTYPES], *kqv_w[NTYPES];
        const float *out_b[NTYPES], *out_w[NTYPES];
        const float *p[ETYPES_N], *skip[NTYPES];
    } L[NL];
    for (int l = 0; l < NL; ++l) {
        for (int e = 0; e < ETYPES_N; ++e) L[l].a_k[e] = F(pi++);
        for (int e = 0; e < ETYPES_N; ++e) L[l].a_v[e] = F(pi++);
        for (int t = 0; t < NTYPES; ++t) { L[l].kqv_b[t] = F(pi++); L[l].kqv_w[t] = F(pi++); }
        for (int t = 0; t < NTYPES; ++t) { L[l].out_b[t] = F(pi++); L[l].out_w[t] = F(pi++); }
        for (int e = 0; e < ETYPES_N; ++e) L[l].p[e] = F(pi++);
        for (int t = 0; t < NTYPES; ++t) L[l].skip[t] = F(pi++);
    }
    const float *proj_b[NTYPES], *proj_w[NTYPES];
    for (int t = 0; t < NTYPES; ++t) { proj_b[t] = F(pi++); proj_w[t] = F(pi++); }
    const float* skipp_b = F(pi++); const float* skipp_w = F(pi++);

    // ---- workspace layout (floats; every region a multiple of 64 floats)
    float* ws = (float*)d_ws;
    size_t off = 0;
    float* h0     = ws + off; off += (size_t)NTOTP * 128;
    float* h1     = ws + off; off += (size_t)NTOTP * 128;
    float* kqvb   = ws + off; off += (size_t)NTOTP * 384;
    float* keAll  = ws + off; off += (size_t)KETOT * 128;
    float* veAll  = ws + off; off += (size_t)KETOT * 128;
    float* alphaB = ws + off; off += (size_t)ETOT * 8;
    int*   nmaxI  = (int*)(ws + off); off += (size_t)NTOTP * 8;
    float* nsum   = ws + off; off += (size_t)NTOTP * 8;
    float* agg    = ws + off; off += (size_t)NTOTP * 128;
    float* hid    = ws + off; off += (size_t)NDOC * 128;
    _Float16* Ah  = (_Float16*)(ws + off); off += (size_t)AH_MAX / 2;
    _Float16* Wt  = (_Float16*)(ws + off); off += (size_t)WT_MAX / 2;

    int keOff[ETYPES_N], aOff[ETYPES_N];
    { int k = 0, a = 0;
      for (int e = 0; e < ETYPES_N; ++e) {
          keOff[e] = k; k += NT_CNT[ET_SRC[e]];
          aOff[e]  = a; a += ET_CNT[e];
      } }

    // conv A + conv W + WMMA GEMM (all padded; M%16==0, K%32==0, N%64==0)
    auto gemm = [&](const float* A, int Mreal, int Ksrc, const float* Wsrc,
                    const float* b, float* C, int Mpad, int N, int Kpad, int ldc,
                    int mode, const float* skipw, const float* hprev) {
        hipLaunchKernelGGL(conv_a_kernel, dim3((Mpad * Kpad + 255) / 256), dim3(256),
                           0, stream, A, Ah, Mreal, Mpad, Ksrc, Kpad, mode == 2 ? 1 : 0);
        hipLaunchKernelGGL(conv_w_kernel, dim3((N * Kpad + 255) / 256), dim3(256),
                           0, stream, Wsrc, Wt, Ksrc, Kpad, N);
        dim3 g((unsigned)(N / 64), (unsigned)((Mpad / 16 + 3) / 4));
        hipLaunchKernelGGL(gemm_wmma_kernel, g, dim3(128), 0, stream,
                           Ah, Wt, b, C, Mpad, N, Kpad, ldc, mode, skipw, hprev);
    };

    // ---- input projection + ReLU
    for (int t = 0; t < NTYPES; ++t)
        gemm(F(NT_XIDX[t]), NT_CNT[t], NT_DIM[t], proj_w[t], proj_b[t],
             h0 + (size_t)NT_OFFP[t] * 128, NT_PAD[t], 128, NT_KPAD[t], 128, 1,
             nullptr, nullptr);

    float* hcur = h0; float* hnext = h1;
    for (int l = 0; l < NL; ++l) {
        // fused K/Q/V projection per node type (N=384)
        for (int t = 0; t < NTYPES; ++t)
            gemm(hcur + (size_t)NT_OFFP[t] * 128, NT_PAD[t], 128,
                 L[l].kqv_w[t], L[l].kqv_b[t],
                 kqvb + (size_t)NT_OFFP[t] * 384, NT_PAD[t], 384, 128, 384, 0,
                 nullptr, nullptr);
        // per-edge-type relation transforms
        for (int e = 0; e < ETYPES_N; ++e) {
            int ns = NT_CNT[ET_SRC[e]];
            hipLaunchKernelGGL(kvrel_kernel, dim3((ns * 128 + 255) / 256), dim3(256), 0, stream,
                               kqvb, NT_OFFP[ET_SRC[e]], L[l].a_k[e], L[l].a_v[e],
                               keAll + (size_t)keOff[e] * 128, veAll + (size_t)keOff[e] * 128, ns);
        }
        hipLaunchKernelGGL(init_nodes_kernel, dim3((NTOTP * 128 + 255) / 256), dim3(256),
                           0, stream, nmaxI, nsum, agg);
        for (int e = 0; e < ETYPES_N; ++e)
            hipLaunchKernelGGL(edge_pass1_kernel, dim3((ET_CNT[e] * 8 + 255) / 256), dim3(256),
                               0, stream, (const int*)d_in[ET_EIIDX[e]], ET_CNT[e],
                               kqvb, NT_OFFP[ET_DST[e]], keAll + (size_t)keOff[e] * 128,
                               L[l].p[e], alphaB + (size_t)aOff[e] * 8, nmaxI);
        for (int e = 0; e < ETYPES_N; ++e)
            hipLaunchKernelGGL(edge_pass2_kernel, dim3((ET_CNT[e] * 8 + 255) / 256), dim3(256),
                               0, stream, (const int*)d_in[ET_EIIDX[e]], ET_CNT[e],
                               NT_OFFP[ET_DST[e]], alphaB + (size_t)aOff[e] * 8, nmaxI, nsum);
        for (int e = 0; e < ETYPES_N; ++e)
            hipLaunchKernelGGL(edge_pass3_kernel, dim3((ET_CNT[e] + 7) / 8), dim3(256),
                               0, stream, (const int*)d_in[ET_EIIDX[e]], ET_CNT[e],
                               NT_OFFP[ET_DST[e]], veAll + (size_t)keOff[e] * 128,
                               alphaB + (size_t)aOff[e] * 8, nsum, agg);
        // out-projection: GELU fused into A-conversion; sigmoid-skip mix + ReLU
        for (int t = 0; t < NTYPES; ++t)
            gemm(agg + (size_t)NT_OFFP[t] * 128, NT_PAD[t], 128,
                 L[l].out_w[t], L[l].out_b[t],
                 hnext + (size_t)NT_OFFP[t] * 128, NT_PAD[t], 128, 128, 128, 2,
                 L[l].skip[t], hcur + (size_t)NT_OFFP[t] * 128);
        float* tmp = hcur; hcur = hnext; hnext = tmp;
    }

    // ---- head: doc = [h_doc | relu(x_doc @ skip_proj)], cls1 ReLU, cls2
    float* logits = (float*)d_out;
    float* docbuf = logits + (size_t)NDOC * 2;      // second tuple output, in place
    hipLaunchKernelGGL(copy_doc_kernel, dim3((NDOC * 128 + 255) / 256), dim3(256),
                       0, stream, hcur, docbuf);
    gemm(F(0), NDOC, 768, skipp_w, skipp_b, docbuf + 128, NDOC, 128, 768, 256, 1,
         nullptr, nullptr);
    gemm(docbuf, NDOC, 256, cls1_w, cls1_b, hid, NDOC, 128, 256, 128, 1,
         nullptr, nullptr);
    hipLaunchKernelGGL(cls2_kernel, dim3((NDOC * 2 + 255) / 256), dim3(256),
                       0, stream, hid, cls2_w, cls2_b, logits);
}
